// GCM_91285234909213
// MI455X (gfx1250) — matrix operand order repaired and checked by reading, not compiled, not run on hardware
//
#include <hip/hip_runtime.h>
#include <hip/hip_bf16.h>
#include <math.h>

// ---------------------------------------------------------------------------
// Graph-matching network on MI455X (gfx1250, wave32, WMMA).
// GEMMs: v_wmma_f32_16x16x32_f16, f32 accum. LDS tiles staged in WMMA
// fragment-major / transposed layouts; fills are pure 16B copies issued as
// GLOBAL_LOAD_ASYNC_TO_LDS_B128 (ASYNCcnt) with double-buffered ping-pong.
// ---------------------------------------------------------------------------

typedef __attribute__((ext_vector_type(16))) _Float16 v16h;
typedef __attribute__((ext_vector_type(8)))  float    v8f;

#define DF      128
#define NPG     512
#define NG      64
#define NNODES  (NG * NPG)
#define NLAYERS 4

union U16h { uint4 q[2]; v16h v; };
union U8h  { _Float16 h[8]; uint4 q; };
union U4h  { _Float16 h[4]; uint2 q; };

__device__ inline v16h frag_ld(const _Float16* p) {  // 32B -> 2x ds_load_b128
    U16h u;
    u.q[0] = *(const uint4*)(p);
    u.q[1] = *(const uint4*)(p + 8);
    return u.v;
}

// Async 16B global -> LDS copy (per-lane gather/scatter, tracked by ASYNCcnt).
// Generic LDS pointer low 32 bits == wave-relative LDS byte offset (ISA 10.2).
__device__ inline void async_copy_b128(void* lds, const void* g) {
    unsigned l = (unsigned)(uintptr_t)lds;
    asm volatile("global_load_async_to_lds_b128 %0, %1, off"
                 :: "v"(l), "v"(g) : "memory");
}
__device__ inline void wait_async() {
    asm volatile("s_wait_asynccnt 0" ::: "memory");
}

#define WMMA(A, B, C) __builtin_amdgcn_wmma_f32_16x16x32_f16(false, A, false, B, (short)0, C, false, false)

// Fragment-major A mapping: row m (0..127), k-group g (0..3): the 8 halves land
// contiguously at aF[m>>4][(m&15)|((g&2)<<3)][(g&1)?8:0]   (k = 16*half + s)
#define AF_DST(BI, m, g) (&aF[BI][(m) >> 4][((m) & 15) | (((g) & 2) << 3)][((g) & 1) ? 8 : 0])

// ---------------------------------------------------------------------------
// Elementwise prep kernels
// ---------------------------------------------------------------------------
__global__ void cvt_kernel(const float* __restrict__ in, _Float16* __restrict__ out) {
    int i = (blockIdx.x * blockDim.x + threadIdx.x) * 4;
    float4 f = *(const float4*)(in + i);
    U4h u;
    u.h[0] = (_Float16)f.x; u.h[1] = (_Float16)f.y;
    u.h[2] = (_Float16)f.z; u.h[3] = (_Float16)f.w;
    *(uint2*)(out + i) = u.q;
}

// wT[l][n][k] = (k<128 ? Wrel[l][k][n] : Wroot[l][k-128][n])  (f16, k-contig)
__global__ void wprep_kernel(const float* __restrict__ Wrel,
                             const float* __restrict__ Wroot,
                             _Float16* __restrict__ wT) {
    int idx = blockIdx.x * blockDim.x + threadIdx.x;      // L*128*256
    int k = idx & 255, n = (idx >> 8) & 127, l = idx >> 15;
    float v = (k < 128) ? Wrel[((size_t)l * 128 + k) * 128 + n]
                        : Wroot[((size_t)l * 128 + (k - 128)) * 128 + n];
    wT[idx] = (_Float16)v;
}

// featT[b][f][node] = in[b*NPG+node][f]  (f16 32x32 LDS tile transpose)
__global__ void __launch_bounds__(256)
transpose_kernel(const _Float16* __restrict__ in, _Float16* __restrict__ outT) {
    __shared__ _Float16 tile[32][33];
    int b = blockIdx.z, n0 = blockIdx.x * 32, f0 = blockIdx.y * 32;
    int tx = threadIdx.x & 31, ty = threadIdx.x >> 5;     // 32 x 8
    const _Float16* ib = in + (size_t)b * NPG * 128;
#pragma unroll
    for (int i = 0; i < 4; ++i)
        tile[ty + i * 8][tx] = ib[(size_t)(n0 + ty + i * 8) * 128 + f0 + tx];
    __syncthreads();
    _Float16* ob = outT + (size_t)b * 128 * NPG;
#pragma unroll
    for (int i = 0; i < 4; ++i)
        ob[(size_t)(f0 + ty + i * 8) * NPG + n0 + tx] = tile[tx][ty + i * 8];
}

// ---------------------------------------------------------------------------
// Edge scatter: one wave per edge; gathers f16 node rows (halved traffic).
// ---------------------------------------------------------------------------
__global__ void scatter_kernel(const _Float16* __restrict__ xh,
                               const float* __restrict__ eattr,
                               const int*   __restrict__ eidx,   // [2, E]
                               float* __restrict__ agg,
                               float* __restrict__ cnt,
                               int E) {
    int t = blockIdx.x * blockDim.x + threadIdx.x;
    int e = t >> 5, lane = t & 31;
    if (e >= E) return;
    int s = eidx[e];
    int d = eidx[E + e];
    int f0 = lane * 4;
    const _Float16* xr = xh + (size_t)s * DF;
    const float*    er = eattr + (size_t)e * DF;
    float*          ar = agg + (size_t)d * DF;
#pragma unroll
    for (int j = 0; j < 4; ++j)
        atomicAdd(&ar[f0 + j], er[f0 + j] * (float)xr[f0 + j]);
    if (lane == 0) atomicAdd(&cnt[d], 1.0f);
}

__global__ void normalize_kernel(const float* __restrict__ agg,
                                 const float* __restrict__ cnt,
                                 _Float16* __restrict__ aggh) {
    int i = blockIdx.x * blockDim.x + threadIdx.x;
    float c = cnt[i >> 7];
    aggh[i] = (_Float16)(agg[i] / fmaxf(c, 1.0f));
}

// ---------------------------------------------------------------------------
// Fused GCM linear: outh = relu([aggh | xh] . wT^T + brel)  (f16 out)
// M=NNODES, N=128, K=256. Block 128 thr (4 waves), tile 128x64, BK=32,
// double-buffered async LDS fills. 8 WMMA / wave / K-step.
// ---------------------------------------------------------------------------
__global__ void __launch_bounds__(128)
lin_gemm_kernel(const _Float16* __restrict__ aggh, const _Float16* __restrict__ xh,
                const _Float16* __restrict__ wT,   // [128][256]
                const float* __restrict__ brel, _Float16* __restrict__ outh) {
    __shared__ _Float16 aF[2][8][32][24];
    __shared__ _Float16 bT[2][64][40];
    const int tid = threadIdx.x, wave = tid >> 5, lane = tid & 31;
    const int m0 = blockIdx.x * 128, n0 = blockIdx.y * 64;
    const v8f vz = {0, 0, 0, 0, 0, 0, 0, 0};
    v8f acc[2][4] = {{vz, vz, vz, vz}, {vz, vz, vz, vz}};

#define LIN_FILL(BI, KTI) do {                                                   \
        int kg0_ = (KTI) * 32;                                                   \
        for (int i_ = tid; i_ < 512; i_ += 128) {                                \
            int m_ = i_ >> 2, g_ = i_ & 3;                                       \
            int row_ = m0 + m_, kg_ = kg0_ + g_ * 8;                             \
            const _Float16* src_ = (kg_ < 128)                                   \
                ? (aggh + (size_t)row_ * 128 + kg_)                              \
                : (xh + (size_t)row_ * 128 + kg_ - 128);                         \
            async_copy_b128(AF_DST(BI, m_, g_), src_);                           \
        }                                                                        \
        for (int i_ = tid; i_ < 256; i_ += 128) {                                \
            int n_ = i_ >> 2, c_ = i_ & 3;                                       \
            async_copy_b128(&bT[BI][n_][c_ * 8],                                 \
                            wT + (size_t)(n0 + n_) * 256 + kg0_ + c_ * 8);       \
        }                                                                        \
    } while (0)

    LIN_FILL(0, 0);
    wait_async();
    __syncthreads();
    for (int kt = 0; kt < 8; ++kt) {
        int cb = kt & 1;
        if (kt + 1 < 8) LIN_FILL(cb ^ 1, kt + 1);
        v16h a0 = frag_ld(&aF[cb][wave * 2][lane][0]);
        v16h a1 = frag_ld(&aF[cb][wave * 2 + 1][lane][0]);
#pragma unroll
        for (int j = 0; j < 4; ++j) {
            v16h bf = frag_ld(&bT[cb][(j << 4) + (lane & 15)][(lane >> 4) * 16]);
            acc[0][j] = WMMA(a0, bf, acc[0][j]);
            acc[1][j] = WMMA(a1, bf, acc[1][j]);
        }
        wait_async();
        __syncthreads();
    }
#undef LIN_FILL
    int cn = lane & 15, mb = (lane >> 4) * 8;
#pragma unroll
    for (int s = 0; s < 2; ++s)
#pragma unroll
        for (int j = 0; j < 4; ++j) {
            int gn = n0 + (j << 4) + cn;
            float bias = brel[gn];
#pragma unroll
            for (int r = 0; r < 8; ++r) {
                int gm = m0 + wave * 32 + s * 16 + mb + r;
                outh[(size_t)gm * 128 + gn] = (_Float16)fmaxf(acc[s][j][r] + bias, 0.0f);
            }
        }
}

// ---------------------------------------------------------------------------
// sim[b] = S_b . T_b^T  (M=512, N=512, K=128 per graph), f16 in, f32 out.
// ---------------------------------------------------------------------------
__global__ void __launch_bounds__(128)
sim_kernel(const _Float16* __restrict__ sh, const _Float16* __restrict__ th,
           float* __restrict__ sim) {
    __shared__ _Float16 aF[2][8][32][24];
    __shared__ _Float16 bT[2][64][40];
    const int tid = threadIdx.x, wave = tid >> 5, lane = tid & 31;
    const int m0 = blockIdx.x * 128, n0 = blockIdx.y * 64, b = blockIdx.z;
    const _Float16* sb = sh + (size_t)b * NPG * 128;
    const _Float16* tb = th + (size_t)b * NPG * 128;
    float* simb = sim + (size_t)b * NPG * NPG;
    const v8f vz = {0, 0, 0, 0, 0, 0, 0, 0};
    v8f acc[2][4] = {{vz, vz, vz, vz}, {vz, vz, vz, vz}};

#define SIM_FILL(BI, KTI) do {                                                   \
        int kg0_ = (KTI) * 32;                                                   \
        for (int i_ = tid; i_ < 512; i_ += 128) {                                \
            int m_ = i_ >> 2, g_ = i_ & 3;                                       \
            async_copy_b128(AF_DST(BI, m_, g_),                                  \
                            sb + (size_t)(m0 + m_) * 128 + kg0_ + g_ * 8);       \
        }                                                                        \
        for (int i_ = tid; i_ < 256; i_ += 128) {                                \
            int n_ = i_ >> 2, c_ = i_ & 3;                                       \
            async_copy_b128(&bT[BI][n_][c_ * 8],                                 \
                            tb + (size_t)(n0 + n_) * 128 + kg0_ + c_ * 8);       \
        }                                                                        \
    } while (0)

    SIM_FILL(0, 0);
    wait_async();
    __syncthreads();
    for (int kt = 0; kt < 4; ++kt) {
        int cb = kt & 1;
        if (kt + 1 < 4) SIM_FILL(cb ^ 1, kt + 1);
        v16h a0 = frag_ld(&aF[cb][wave * 2][lane][0]);
        v16h a1 = frag_ld(&aF[cb][wave * 2 + 1][lane][0]);
#pragma unroll
        for (int j = 0; j < 4; ++j) {
            v16h bf = frag_ld(&bT[cb][(j << 4) + (lane & 15)][(lane >> 4) * 16]);
            acc[0][j] = WMMA(a0, bf, acc[0][j]);
            acc[1][j] = WMMA(a1, bf, acc[1][j]);
        }
        wait_async();
        __syncthreads();
    }
#undef SIM_FILL
    int cn = lane & 15, mb = (lane >> 4) * 8;
#pragma unroll
    for (int s2 = 0; s2 < 2; ++s2)
#pragma unroll
        for (int j = 0; j < 4; ++j) {
            int gn = n0 + (j << 4) + cn;
#pragma unroll
            for (int r = 0; r < 8; ++r) {
                int gm = m0 + wave * 32 + s2 * 16 + mb + r;
                simb[(size_t)gm * NPG + gn] = acc[s2][j][r];
            }
        }
}

// ---------------------------------------------------------------------------
// Softmax stats: max and 1/sum(exp(.-max)) along rows (axis=0) / cols (axis=1)
// ---------------------------------------------------------------------------
__global__ void stats_kernel(const float* __restrict__ sim,
                             float* __restrict__ mx, float* __restrict__ rs,
                             int axis) {
    int i = blockIdx.x, b = blockIdx.y, tid = threadIdx.x;
    const float* simb = sim + (size_t)b * NPG * NPG;
    __shared__ float red[256];
    float m = -3.0e38f;
    for (int j = tid; j < NPG; j += 256) {
        float v = (axis == 0) ? simb[(size_t)i * NPG + j] : simb[(size_t)j * NPG + i];
        m = fmaxf(m, v);
    }
    red[tid] = m; __syncthreads();
    for (int s2 = 128; s2 > 0; s2 >>= 1) {
        if (tid < s2) red[tid] = fmaxf(red[tid], red[tid + s2]);
        __syncthreads();
    }
    m = red[0]; __syncthreads();
    float sum = 0.0f;
    for (int j = tid; j < NPG; j += 256) {
        float v = (axis == 0) ? simb[(size_t)i * NPG + j] : simb[(size_t)j * NPG + i];
        sum += __expf(v - m);
    }
    red[tid] = sum; __syncthreads();
    for (int s2 = 128; s2 > 0; s2 >>= 1) {
        if (tid < s2) red[tid] += red[tid + s2];
        __syncthreads();
    }
    if (tid == 0) { mx[b * NPG + i] = m; rs[b * NPG + i] = 1.0f / red[0]; }
}

// ---------------------------------------------------------------------------
// Softmax-weighted GEMM: out[b] = softmax(sim) . feat[b]   (featT is f16,
// feature-major [b][f][node] so B fills are contiguous async copies).
// transposed=0: A[m,k]=exp(sim[m,k]-rmax[m])*rrcp[m]; =1: sim[k,m]/col stats.
// Writes f32 (pooling) and f16 (next layer) outputs.
// ---------------------------------------------------------------------------
__global__ void __launch_bounds__(128)
attn_gemm_kernel(const float* __restrict__ sim, const float* __restrict__ mx,
                 const float* __restrict__ rs, const _Float16* __restrict__ featT,
                 float* __restrict__ out, _Float16* __restrict__ outh,
                 int transposed) {
    __shared__ _Float16 aF[2][8][32][24];
    __shared__ _Float16 bT[2][64][40];
    const int tid = threadIdx.x, wave = tid >> 5, lane = tid & 31;
    const int m0 = blockIdx.x * 128, n0 = blockIdx.y * 64, b = blockIdx.z;
    const float* simb = sim + (size_t)b * NPG * NPG;
    const _Float16* fT = featT + (size_t)b * 128 * NPG;
    const float* mxb = mx + b * NPG;
    const float* rsb = rs + b * NPG;
    const v8f vz = {0, 0, 0, 0, 0, 0, 0, 0};
    v8f acc[2][4] = {{vz, vz, vz, vz}, {vz, vz, vz, vz}};

#define ATTN_FILL(BI, KTI) do {                                                  \
        int kg0_ = (KTI) * 32;                                                   \
        for (int i_ = tid; i_ < 512; i_ += 128) {                                \
            int m_ = i_ >> 2, g_ = i_ & 3;                                       \
            int row_ = m0 + m_, kg_ = kg0_ + g_ * 8;                             \
            float mv_ = mxb[row_], rv_ = rsb[row_];                              \
            float v_[8];                                                         \
            if (!transposed) {                                                   \
                const float* src_ = simb + (size_t)row_ * NPG + kg_;             \
                float4 f0_ = ((const float4*)src_)[0];                           \
                float4 f1_ = ((const float4*)src_)[1];                           \
                v_[0] = f0_.x; v_[1] = f0_.y; v_[2] = f0_.z; v_[3] = f0_.w;      \
                v_[4] = f1_.x; v_[5] = f1_.y; v_[6] = f1_.z; v_[7] = f1_.w;      \
            } else {                                                             \
                _Pragma("unroll")                                                \
                for (int j_ = 0; j_ < 8; ++j_)                                   \
                    v_[j_] = simb[(size_t)(kg_ + j_) * NPG + row_];              \
            }                                                                    \
            U8h u_;                                                              \
            _Pragma("unroll")                                                    \
            for (int j_ = 0; j_ < 8; ++j_)                                       \
                u_.h[j_] = (_Float16)(__expf(v_[j_] - mv_) * rv_);               \
            *(uint4*)AF_DST(BI, m_, g_) = u_.q;                                  \
        }                                                                        \
        for (int i_ = tid; i_ < 256; i_ += 128) {                                \
            int n_ = i_ >> 2, c_ = i_ & 3;                                       \
            async_copy_b128(&bT[BI][n_][c_ * 8],                                 \
                            fT + (size_t)(n0 + n_) * NPG + kg0_ + c_ * 8);       \
        }                                                                        \
    } while (0)

    ATTN_FILL(0, 0);
    wait_async();
    __syncthreads();
    for (int kt = 0; kt < 16; ++kt) {
        int cb = kt & 1;
        if (kt + 1 < 16) ATTN_FILL(cb ^ 1, kt + 1);
        v16h a0 = frag_ld(&aF[cb][wave * 2][lane][0]);
        v16h a1 = frag_ld(&aF[cb][wave * 2 + 1][lane][0]);
#pragma unroll
        for (int j = 0; j < 4; ++j) {
            v16h bf = frag_ld(&bT[cb][(j << 4) + (lane & 15)][(lane >> 4) * 16]);
            acc[0][j] = WMMA(a0, bf, acc[0][j]);
            acc[1][j] = WMMA(a1, bf, acc[1][j]);
        }
        wait_async();
        __syncthreads();
    }
#undef ATTN_FILL
    int cn = lane & 15, mb = (lane >> 4) * 8;
#pragma unroll
    for (int s2 = 0; s2 < 2; ++s2)
#pragma unroll
        for (int j = 0; j < 4; ++j) {
            int gn = n0 + (j << 4) + cn;
#pragma unroll
            for (int r = 0; r < 8; ++r) {
                int gm = m0 + wave * 32 + s2 * 16 + mb + r;
                size_t idx = ((size_t)b * NPG + gm) * 128 + gn;
                float v = acc[s2][j][r];
                out[idx]  = v;
                outh[idx] = (_Float16)v;
            }
        }
}

// ---------------------------------------------------------------------------
// Per-graph mean pool + cosine similarity. One 128-thread block per graph.
// ---------------------------------------------------------------------------
__global__ void pool_cos_kernel(const float* __restrict__ sx,
                                const float* __restrict__ tx,
                                float* __restrict__ out) {
    int b = blockIdx.x, f = threadIdx.x;
    const float* sb = sx + (size_t)b * NPG * 128;
    const float* tb = tx + (size_t)b * NPG * 128;
    float ss = 0.0f, tt = 0.0f;
    for (int n = 0; n < NPG; ++n) { ss += sb[n * 128 + f]; tt += tb[n * 128 + f]; }
    ss *= (1.0f / NPG); tt *= (1.0f / NPG);
    __shared__ float rd[3][128];
    rd[0][f] = ss * tt; rd[1][f] = ss * ss; rd[2][f] = tt * tt;
    __syncthreads();
    for (int s2 = 64; s2 > 0; s2 >>= 1) {
        if (f < s2) {
            rd[0][f] += rd[0][f + s2];
            rd[1][f] += rd[1][f + s2];
            rd[2][f] += rd[2][f + s2];
        }
        __syncthreads();
    }
    if (f == 0) {
        const float eps = 1e-8f;
        float ns = fmaxf(sqrtf(rd[1][0]), eps);
        float nt = fmaxf(sqrtf(rd[2][0]), eps);
        out[b] = rd[0][0] / (ns * nt);
    }
}

// ---------------------------------------------------------------------------
extern "C" void kernel_launch(void* const* d_in, const int* in_sizes, int n_in,
                              void* d_out, int out_size, void* d_ws, size_t ws_size,
                              hipStream_t stream) {
    const float* src_x  = (const float*)d_in[0];
    const float* tgt_x  = (const float*)d_in[1];
    const float* src_ea = (const float*)d_in[2];
    const float* tgt_ea = (const float*)d_in[3];
    const float* Wrel   = (const float*)d_in[4];
    const float* brel   = (const float*)d_in[5];
    const float* Wroot  = (const float*)d_in[6];
    const int*   src_ei = (const int*)d_in[7];
    const int*   tgt_ei = (const int*)d_in[8];
    float* out = (float*)d_out;

    const int E = in_sizes[2] / DF;

    const size_t featF = (size_t)NNODES * DF * sizeof(float);    // 16 MB
    const size_t featH = (size_t)NNODES * DF * sizeof(_Float16); //  8 MB
    const size_t cntB  = (size_t)NNODES * sizeof(float);
    const size_t statB = (size_t)NG * NPG * sizeof(float);
    char* w = (char*)d_ws;
    float*    cur_s  = (float*)w;    w += featF;
    float*    cur_t  = (float*)w;    w += featF;
    float*    agg_s  = (float*)w;    w += featF;
    float*    agg_t  = (float*)w;    w += featF;
    _Float16* curh_s = (_Float16*)w; w += featH;
    _Float16* curh_t = (_Float16*)w; w += featH;
    _Float16* aggh_s = (_Float16*)w; w += featH;
    _Float16* aggh_t = (_Float16*)w; w += featH;
    _Float16* linh_s = (_Float16*)w; w += featH;
    _Float16* linh_t = (_Float16*)w; w += featH;
    _Float16* fT_s   = (_Float16*)w; w += featH;
    _Float16* fT_t   = (_Float16*)w; w += featH;
    _Float16* wT     = (_Float16*)w; w += (size_t)NLAYERS * 128 * 256 * sizeof(_Float16);
    float*    cnt_s  = (float*)w;    w += cntB;
    float*    cnt_t  = (float*)w;    w += cntB;
    float*    rmax   = (float*)w;    w += statB;
    float*    rrs    = (float*)w;    w += statB;
    float*    cmax   = (float*)w;    w += statB;
    float*    rcs    = (float*)w;    w += statB;
    float*    sim    = (float*)w;    // 64*512*512*4 = 67 MB

    // One-time prep: f16 feature mirrors + fragment-ready transposed weights.
    cvt_kernel<<<(NNODES * DF) / 1024, 256, 0, stream>>>(src_x, curh_s);
    cvt_kernel<<<(NNODES * DF) / 1024, 256, 0, stream>>>(tgt_x, curh_t);
    wprep_kernel<<<(NLAYERS * 128 * 256) / 256, 256, 0, stream>>>(Wrel, Wroot, wT);

    const int scatterBlocks = (E * 32 + 255) / 256;
    const int normBlocks    = (NNODES * DF) / 256;

    for (int l = 0; l < NLAYERS; ++l) {
        const _Float16* wT_l  = wT + (size_t)l * 128 * 256;
        const float*    brl   = brel + (size_t)l * DF;

        hipMemsetAsync(agg_s, 0, featF, stream);
        hipMemsetAsync(agg_t, 0, featF, stream);
        hipMemsetAsync(cnt_s, 0, cntB, stream);
        hipMemsetAsync(cnt_t, 0, cntB, stream);

        scatter_kernel<<<scatterBlocks, 256, 0, stream>>>(curh_s, src_ea, src_ei, agg_s, cnt_s, E);
        scatter_kernel<<<scatterBlocks, 256, 0, stream>>>(curh_t, tgt_ea, tgt_ei, agg_t, cnt_t, E);
        normalize_kernel<<<normBlocks, 256, 0, stream>>>(agg_s, cnt_s, aggh_s);
        normalize_kernel<<<normBlocks, 256, 0, stream>>>(agg_t, cnt_t, aggh_t);

        lin_gemm_kernel<<<dim3(NNODES / 128, 2), 128, 0, stream>>>(aggh_s, curh_s, wT_l, brl, linh_s);
        lin_gemm_kernel<<<dim3(NNODES / 128, 2), 128, 0, stream>>>(aggh_t, curh_t, wT_l, brl, linh_t);

        transpose_kernel<<<dim3(NPG / 32, DF / 32, NG), 256, 0, stream>>>(linh_s, fT_s);
        transpose_kernel<<<dim3(NPG / 32, DF / 32, NG), 256, 0, stream>>>(linh_t, fT_t);

        sim_kernel<<<dim3(NPG / 128, NPG / 64, NG), 128, 0, stream>>>(linh_s, linh_t, sim);

        stats_kernel<<<dim3(NPG, NG), 256, 0, stream>>>(sim, rmax, rrs, 0);
        stats_kernel<<<dim3(NPG, NG), 256, 0, stream>>>(sim, cmax, rcs, 1);

        attn_gemm_kernel<<<dim3(NPG / 128, 2, NG), 128, 0, stream>>>(
            sim, rmax, rrs, fT_t, cur_s, curh_s, 0);
        attn_gemm_kernel<<<dim3(NPG / 128, 2, NG), 128, 0, stream>>>(
            sim, cmax, rcs, fT_s, cur_t, curh_t, 1);
    }

    pool_cos_kernel<<<NG, 128, 0, stream>>>(cur_s, cur_t, out);
    (void)n_in; (void)out_size; (void)ws_size;
}
